// SirenLinear_13529146982441
// MI455X (gfx1250) — compile-verified, matrix operand order checked by reading, not run on hardware
//
#include <hip/hip_runtime.h>

// MI455X (gfx1250, wave32) implementation.
// Kernel 1: SIREN hypernet (18->64->64->1, sin(30x)) over 1M points -> W[1024][1024] in d_ws.
//   Transposed formulation (features = M, points = N) so layer outputs re-enter as B fragments
//   via per-wave LDS staging. Split-bf16 (hi+lo) x 3 WMMA products gives near-fp32 accuracy on
//   v_wmma_f32_16x16x32_bf16 (8x FLOP/instr vs f32 16x16x4 WMMA).
// Kernel 2: out = x @ W^T + bias. Double-buffered K loop fed by GLOBAL_LOAD_ASYNC_TO_LDS_B128
//   (ASYNCcnt path, inline asm) so HBM->LDS tile movement overlaps the WMMA stream; bf16 hi/lo
//   split happens at fragment-build time from fp32 LDS tiles.

typedef __bf16 bf16;
typedef bf16         v16bf  __attribute__((ext_vector_type(16)));
typedef float        v8f    __attribute__((ext_vector_type(8)));
typedef unsigned int u32x4  __attribute__((ext_vector_type(4)));
typedef float        f32x4  __attribute__((ext_vector_type(4)));

#define SIREN_W0 30.0f

// Load one 16x32 bf16 A/B fragment row for this lane.
// CDNA5 16-bit A/B layout: lanes 0-15 hold K = {half..half+7, half+16..half+23} as 8+8
// contiguous bf16 -> two ds_load_b128.  `row` points at the 32-K-wide row for this lane's M/N.
__device__ __forceinline__ v16bf frag_ld(const bf16* row, int half) {
    union { u32x4 q[2]; v16bf v; } u;
    u.q[0] = *(const u32x4*)(row + half);
    u.q[1] = *(const u32x4*)(row + half + 16);
    return u.v;
}

__device__ __forceinline__ v8f wmma_bf16(v16bf a, v16bf b, v8f c) {
    return __builtin_amdgcn_wmma_f32_16x16x32_bf16(false, a, false, b, (short)0, c, false, false);
}

__device__ __forceinline__ void split_bf16(float v, bf16& hi, bf16& lo) {
    hi = (bf16)v;
    lo = (bf16)(v - (float)hi);
}

// Build hi/lo bf16 fragments for this lane directly from an fp32 LDS tile row
// (row stride = 32 floats).  4x ds_load_b128 + cvt; one read feeds both fragments.
__device__ __forceinline__ void frag_cvt(const float* row, int half, v16bf& hi, v16bf& lo) {
    f32x4 a0 = *(const f32x4*)(row + half);
    f32x4 a1 = *(const f32x4*)(row + half + 4);
    f32x4 b0 = *(const f32x4*)(row + half + 16);
    f32x4 b1 = *(const f32x4*)(row + half + 20);
    float v[16] = {a0[0], a0[1], a0[2], a0[3], a1[0], a1[1], a1[2], a1[3],
                   b0[0], b0[1], b0[2], b0[3], b1[0], b1[1], b1[2], b1[3]};
#pragma unroll
    for (int e = 0; e < 16; ++e) {
        bf16 h = (bf16)v[e];
        hi[e] = h;
        lo[e] = (bf16)(v[e] - (float)h);
    }
}

// Async global -> LDS (per-lane 16B), GVS addressing: SADDR(64) + VADDR(32) byte offset.
__device__ __forceinline__ void async_ld_b128(unsigned lds_addr, unsigned voff,
                                              unsigned long long base) {
    asm volatile("global_load_async_to_lds_b128 %0, %1, %2"
                 :: "v"(lds_addr), "v"(voff), "s"(base) : "memory");
}

__device__ __forceinline__ void wait_async0() {
#if defined(__has_builtin) && __has_builtin(__builtin_amdgcn_s_wait_asynccnt)
    __builtin_amdgcn_s_wait_asynccnt(0);
#else
    asm volatile("s_wait_asynccnt 0" ::: "memory");
#endif
}

__device__ __forceinline__ unsigned lds_off(const void* p) {
    return (unsigned)(unsigned long long)p;   // low 32 bits of generic ptr = LDS byte address
}

// ---------------------------------------------------------------------------
// Kernel A: generate W[1024*1024] from embedded coords via the SIREN MLP.
// Block = 256 threads = 8 waves; each wave processes 16 points per tile.
// ---------------------------------------------------------------------------
__global__ __launch_bounds__(256) void siren_hypernet(
    const float* __restrict__ EC,   // [1M x 18]
    const float* __restrict__ W1,   // [64 x 18]
    const float* __restrict__ B1,   // [64]
    const float* __restrict__ W2,   // [64 x 64]
    const float* __restrict__ B2,   // [64]
    const float* __restrict__ W3,   // [64]
    const float* __restrict__ B3,   // [1]
    float* __restrict__ Wout)       // [1024*1024]
{
    __shared__ __align__(16) bf16 sW1h[64 * 32], sW1l[64 * 32];   // K padded 18 -> 32
    __shared__ __align__(16) bf16 sW2h[64 * 64], sW2l[64 * 64];
    __shared__ float sB1[64], sB2[64], sW3[64];
    __shared__ float sB3;
    __shared__ __align__(16) bf16 sEh[8][16 * 32], sEl[8][16 * 32];  // per-wave E^T tile
    __shared__ __align__(16) bf16 sHh[8][16 * 64], sHl[8][16 * 64];  // per-wave hidden tile

    const int t = threadIdx.x;

    // Stage weights once per block, pre-split into bf16 hi/lo.
    for (int i = t; i < 64 * 32; i += 256) {
        int m = i >> 5, k = i & 31;
        float v = (k < 18) ? W1[m * 18 + k] : 0.0f;
        split_bf16(v, sW1h[i], sW1l[i]);
    }
    for (int i = t; i < 64 * 64; i += 256) {
        split_bf16(W2[i], sW2h[i], sW2l[i]);
    }
    if (t < 64) { sB1[t] = B1[t]; sB2[t] = B2[t]; sW3[t] = W3[t]; }
    if (t == 0) sB3 = B3[0];
    __syncthreads();

    const int w    = t >> 5;
    const int lane = t & 31;
    const int ln   = lane & 15;
    const int half = (lane & 16) ? 8 : 0;   // lane-half K offset (also C-layout M offset)

    bf16* eh = sEh[w]; bf16* el = sEl[w];
    bf16* hh = sHh[w]; bf16* hl = sHl[w];

    const v8f zero = {0.f, 0.f, 0.f, 0.f, 0.f, 0.f, 0.f, 0.f};

    const int gw = blockIdx.x * 8 + w;
    const int stride = gridDim.x * 8;
    for (int tile = gw; tile < 65536; tile += stride) {
        const int p0 = tile << 4;   // first point of this 16-point tile

        // Stage E^T tile: [16 points][32 padded k], split hi/lo (wave-private, no barrier).
        for (int s = lane; s < 512; s += 32) {
            int nn = s >> 5, kk = s & 31;
            float v = (kk < 18) ? EC[(p0 + nn) * 18 + kk] : 0.0f;
            split_bf16(v, eh[s], el[s]);
        }

        // ---- Layer 1: D1[64h x 16pts] = W1 @ E^T  (12 WMMAs) ----
        v8f acc[4] = {zero, zero, zero, zero};
        v16bf beh = frag_ld(eh + ln * 32, half);
        v16bf bel = frag_ld(el + ln * 32, half);
#pragma unroll
        for (int mt = 0; mt < 4; ++mt) {
            const int row = (mt * 16 + ln) * 32;
            v16bf ah = frag_ld(sW1h + row, half);
            v16bf al = frag_ld(sW1l + row, half);
            acc[mt] = wmma_bf16(ah, beh, acc[mt]);
            acc[mt] = wmma_bf16(ah, bel, acc[mt]);
            acc[mt] = wmma_bf16(al, beh, acc[mt]);
        }

        // sin(30*(.+b1)), split, pack into [point][k] so it reloads as a B fragment.
        // C layout: vgpr r holds (m = mt*16 + r + half, n = ln) -> 8 consecutive k per store.
#pragma unroll
        for (int mt = 0; mt < 4; ++mt) {
            union { u32x4 q; bf16 b[8]; } ph, pl;
#pragma unroll
            for (int r = 0; r < 8; ++r) {
                int m = mt * 16 + r + half;
                float u = __sinf(SIREN_W0 * (acc[mt][r] + sB1[m]));
                split_bf16(u, ph.b[r], pl.b[r]);
            }
            *(u32x4*)(hh + ln * 64 + mt * 16 + half) = ph.q;
            *(u32x4*)(hl + ln * 64 + mt * 16 + half) = pl.q;
        }

        // ---- Layer 2: D2[64h x 16pts] = W2 @ H1  (24 WMMAs) ----
        v8f acc2[4] = {zero, zero, zero, zero};
#pragma unroll
        for (int c = 0; c < 2; ++c) {
            v16bf bh = frag_ld(hh + ln * 64 + c * 32, half);
            v16bf bl = frag_ld(hl + ln * 64 + c * 32, half);
#pragma unroll
            for (int mt = 0; mt < 4; ++mt) {
                const int row = (mt * 16 + ln) * 64 + c * 32;
                v16bf ah = frag_ld(sW2h + row, half);
                v16bf al = frag_ld(sW2l + row, half);
                acc2[mt] = wmma_bf16(ah, bh, acc2[mt]);
                acc2[mt] = wmma_bf16(ah, bl, acc2[mt]);
                acc2[mt] = wmma_bf16(al, bh, acc2[mt]);
            }
        }

        // ---- Layer 3: per-point dot(h2, W3) + b3.  Each lane holds 32 k-values of its
        // point (n = ln); lanes l and l^16 together cover all 64 k -> one xor-16 reduce.
        float part = 0.0f;
#pragma unroll
        for (int mt = 0; mt < 4; ++mt)
#pragma unroll
            for (int r = 0; r < 8; ++r) {
                int m = mt * 16 + r + half;
                float u = __sinf(SIREN_W0 * (acc2[mt][r] + sB2[m]));
                part = fmaf(u, sW3[m], part);
            }
        part += __shfl_xor(part, 16);
        if (lane < 16) Wout[p0 + ln] = part + sB3;
    }
}

// ---------------------------------------------------------------------------
// Kernel B: out[8192x1024] = x @ W^T + bias.
// Block = 256 threads (8 waves) -> 128(M) x 64(N) tile; wave owns 32x32 (4 C tiles).
// K walked in chunks of 32; fp32 tiles streamed into double-buffered LDS with
// GLOBAL_LOAD_ASYNC_TO_LDS_B128 (ASYNCcnt), bf16 hi/lo split at fragment build.
// ---------------------------------------------------------------------------
__global__ __launch_bounds__(256) void siren_apply_gemm(
    const float* __restrict__ X,    // [8192 x 1024]
    const float* __restrict__ Wg,   // [1024 x 1024] generated weights
    const float* __restrict__ bias, // [1024]
    float* __restrict__ out)        // [8192 x 1024]
{
    __shared__ __align__(16) float sX[2][128 * 32];  // 16 KB x2
    __shared__ __align__(16) float sW[2][64 * 32];   //  8 KB x2

    const int t    = threadIdx.x;
    const int w    = t >> 5;
    const int lane = t & 31;
    const int ln   = lane & 15;
    const int half = (lane & 16) ? 8 : 0;
    const int wm   = w & 3;        // i sub-tile (32 rows)
    const int wn   = w >> 2;       // j sub-tile (32 cols)
    const int ibase = (blockIdx.x >> 4) * 128;
    const int jbase = (blockIdx.x & 15) * 64;

    const unsigned long long xb = (unsigned long long)X;
    const unsigned long long wb = (unsigned long long)Wg;

    const v8f zero = {0.f, 0.f, 0.f, 0.f, 0.f, 0.f, 0.f, 0.f};
    v8f acc[2][2] = {{zero, zero}, {zero, zero}};

    // Issue async staging of one 128x32 X tile + 64x32 W tile into buffer `b`.
    auto stage = [&](int b, int k0) {
#pragma unroll
        for (int rep = 0; rep < 4; ++rep) {             // X: 1024 x 16B chunks
            int q = t + rep * 256;
            int i = q >> 3, kq = (q & 7) << 2;
            unsigned voff = (unsigned)(((ibase + i) * 1024 + k0 + kq) * 4);
            async_ld_b128(lds_off(&sX[b][i * 32 + kq]), voff, xb);
        }
#pragma unroll
        for (int rep = 0; rep < 2; ++rep) {             // W: 512 x 16B chunks
            int q = t + rep * 256;
            int j = q >> 3, kq = (q & 7) << 2;
            unsigned voff = (unsigned)(((jbase + j) * 1024 + k0 + kq) * 4);
            async_ld_b128(lds_off(&sW[b][j * 32 + kq]), voff, wb);
        }
    };

    int buf = 0;
    stage(0, 0);
    wait_async0();
    __syncthreads();

    for (int k0 = 0; k0 < 1024; k0 += 32) {
        if (k0 + 32 < 1024) stage(buf ^ 1, k0 + 32);   // overlap next tile with compute

        v16bf bh[2], bl[2];
#pragma unroll
        for (int ns = 0; ns < 2; ++ns)
            frag_cvt(&sW[buf][(wn * 32 + ns * 16 + ln) * 32], half, bh[ns], bl[ns]);
#pragma unroll
        for (int ms = 0; ms < 2; ++ms) {
            v16bf ah, al;
            frag_cvt(&sX[buf][(wm * 32 + ms * 16 + ln) * 32], half, ah, al);
#pragma unroll
            for (int ns = 0; ns < 2; ++ns) {
                acc[ms][ns] = wmma_bf16(ah, bh[ns], acc[ms][ns]);
                acc[ms][ns] = wmma_bf16(ah, bl[ns], acc[ms][ns]);
                acc[ms][ns] = wmma_bf16(al, bh[ns], acc[ms][ns]);
            }
        }

        wait_async0();      // next tile landed in LDS
        __syncthreads();    // everyone done reading `buf`, `buf^1` visible to all
        buf ^= 1;
    }

    // Epilogue: + bias, store.  C layout: vgpr r -> row (r + half), col = ln.
#pragma unroll
    for (int ms = 0; ms < 2; ++ms)
#pragma unroll
        for (int ns = 0; ns < 2; ++ns) {
            const int j  = jbase + wn * 32 + ns * 16 + ln;
            const float bv = bias[j];
            const int rbase = ibase + wm * 32 + ms * 16 + half;
#pragma unroll
            for (int r = 0; r < 8; ++r)
                out[(rbase + r) * 1024 + j] = acc[ms][ns][r] + bv;
        }
}

extern "C" void kernel_launch(void* const* d_in, const int* in_sizes, int n_in,
                              void* d_out, int out_size, void* d_ws, size_t ws_size,
                              hipStream_t stream) {
    const float* x    = (const float*)d_in[0];
    const float* EC   = (const float*)d_in[1];
    const float* W1   = (const float*)d_in[2];
    const float* b1   = (const float*)d_in[3];
    const float* W2   = (const float*)d_in[4];
    const float* b2   = (const float*)d_in[5];
    const float* W3   = (const float*)d_in[6];
    const float* b3   = (const float*)d_in[7];
    const float* bias = (const float*)d_in[8];
    float* out  = (float*)d_out;
    float* Wgen = (float*)d_ws;     // 1024*1024 fp32 = 4 MB scratch

    (void)in_sizes; (void)n_in; (void)out_size; (void)ws_size;

    // 1024 blocks x 8 waves -> each wave runs 8 sixteen-point tiles.
    siren_hypernet<<<1024, 256, 0, stream>>>(EC, W1, b1, W2, b2, W3, b3, Wgen);
    // 64 (M) x 16 (N) tile grid.
    siren_apply_gemm<<<1024, 256, 0, stream>>>(x, Wgen, bias, out);
}
